// Attention_52218212385391
// MI455X (gfx1250) — matrix operand order repaired and checked by reading, not compiled, not run on hardware
//
#include <hip/hip_runtime.h>
#include <hip/hip_bf16.h>

// ---------------------------------------------------------------------------
// Types for CDNA5 (wave32): v_wmma_f32_16x16x32_bf16 + TDM descriptors
// ---------------------------------------------------------------------------
typedef __attribute__((ext_vector_type(16))) __bf16        v16bf;
typedef __attribute__((ext_vector_type(8)))  float         v8f;
typedef __attribute__((ext_vector_type(4)))  unsigned int  v4ui;
typedef __attribute__((ext_vector_type(8)))  int           v8i;
typedef __attribute__((ext_vector_type(4)))  int           v4i;

union BF16Frag {
    v16bf          bf;
    unsigned short us[16];
};

__device__ __forceinline__ unsigned short f2bf(float f) {
    unsigned int u = __float_as_uint(f);
    u += 0x7FFFu + ((u >> 16) & 1u);        // round-to-nearest-even
    return (unsigned short)(u >> 16);
}

__device__ __forceinline__ int iclampi(int v, int lo, int hi) {
    return v < lo ? lo : (v > hi ? hi : v);
}

// ---------------------------------------------------------------------------
// Problem constants
// ---------------------------------------------------------------------------
#define BATCH   128
#define NTOK    196
#define CDIM    768
#define GDIM    8
#define CGDIM   96
#define NHEAD   12
#define DHEAD   64
#define MTOK    (BATCH * NTOK)            // 25088
#define S1ELEM  ((size_t)MTOK * CDIM)     // token-major feature map elems
#define WELEM   ((size_t)CDIM * CDIM)

#if __has_builtin(__builtin_amdgcn_tensor_load_to_lds)
#define HAVE_TDM 1
#else
#define HAVE_TDM 0
#endif

// ---------------------------------------------------------------------------
// TDM 2D tile load: 128 rows x 32 bf16 cols, row stride = K elems.
// D# per CDNA5 ISA ch.8: group0 {count=1, lds_addr, global_addr(57b), type=2},
// group1 {data_size=2B, tensor_dim0=K, tensor_dim1=128, tile=32x128,
//         tensor_dim0_stride=K}; groups 2/3 zero (2-D tensor).
// ---------------------------------------------------------------------------
#if HAVE_TDM
__device__ __forceinline__ void tdm_load_tile_128x32(const void* gptr,
                                                     const void* ldsptr,
                                                     unsigned k_elems) {
    unsigned long long ga = (unsigned long long)(size_t)gptr;
    unsigned lds_off = (unsigned)(size_t)ldsptr;           // LDS aperture: low 32b
    v4ui g0 = { 1u,                                        // count=1, user mode
                lds_off,
                (unsigned)ga,
                (unsigned)((ga >> 32) & 0x01FFFFFFull) | 0x80000000u }; // type=2
    v8i g1 = { (int)(1u << 16),                // data_size=1 (2 bytes), mask=0
               (int)((k_elems & 0xFFFFu) << 16),  // tensor_dim0[15:0] @bit48
               (int)(128u << 16),                 // tensor_dim1=128 @bit80
               (int)(32u << 16),                  // tile_dim0=32    @bit112
               (int)128,                          // tile_dim1=128   @bit128
               (int)k_elems,                      // tensor_dim0_stride @bit160
               0, 0 };
    v4i z4 = { 0, 0, 0, 0 };
#if defined(__clang_major__) && __clang_major__ >= 23
    v8i z8 = { 0, 0, 0, 0, 0, 0, 0, 0 };
    __builtin_amdgcn_tensor_load_to_lds(g0, g1, z4, z4, z8, 0);
#else
    __builtin_amdgcn_tensor_load_to_lds(g0, g1, z4, z4, 0);
#endif
}
#endif

// ---------------------------------------------------------------------------
// fp32 -> packed bf16 conversion (one-time per operand; keeps the rounding
// VALU work out of the GEMM inner loop)
// ---------------------------------------------------------------------------
__global__ __launch_bounds__(256)
void to_bf16_kernel(const float* __restrict__ in, unsigned int* __restrict__ out,
                    long n2 /* = n/2 */) {
    long i = (long)blockIdx.x * blockDim.x + threadIdx.x;
    if (i < n2) {
        float a = in[2 * i], b = in[2 * i + 1];
        out[i] = (unsigned)f2bf(a) | ((unsigned)f2bf(b) << 16);
    }
}

// ---------------------------------------------------------------------------
// GEMM: Out[M,N] = A[M,K] * W[N,K]^T + bias   (bf16 in, fp32 accumulate/out)
// Block tile 128x128, 8 waves, each wave 32x64 (2x4 WMMA tiles), K-step 32.
// Global->LDS via TDM tensor_load_to_lds, double buffered on TENSORcnt.
// Requires M%128==0, N%128==0, K%32==0.
// ---------------------------------------------------------------------------
#define GTM 128
#define GTN 128
#define GTK 32

__global__ __launch_bounds__(256)
void gemm_bias_bf16(const unsigned short* __restrict__ A_bf,
                    const unsigned short* __restrict__ W_bf,
                    const float* __restrict__ bias,
                    float* __restrict__ Out,
                    int M, int N, int K)
{
    __shared__ unsigned short lA[2][GTM * GTK];
    __shared__ unsigned short lB[2][GTN * GTK];

    const int bm   = blockIdx.x * GTM;
    const int bn   = blockIdx.y * GTN;
    const int tid  = threadIdx.x;
    const int wave = tid >> 5;
    const int lane = tid & 31;
    const int l15  = lane & 15;
    const int lhi  = lane >> 4;
    const int wm   = (wave & 3) * 32;    // 4 wave-rows  -> 128 M
    const int wn   = (wave >> 2) * 64;   // 2 wave-cols  -> 128 N
    const int steps = K / GTK;

    v8f acc[2][4] = {};

#if HAVE_TDM
    if (wave == 0) {   // one wave drives the Tensor Data Mover
        tdm_load_tile_128x32(A_bf + (size_t)bm * K, &lA[0][0], (unsigned)K);
        tdm_load_tile_128x32(W_bf + (size_t)bn * K, &lB[0][0], (unsigned)K);
    }
#endif

    for (int kt = 0; kt < steps; ++kt) {
        const int cur = kt & 1;
#if HAVE_TDM
        if (wave == 0) {
            if (kt + 1 < steps) {      // keep next slab in flight
                const int nxt = cur ^ 1;
                tdm_load_tile_128x32(A_bf + (size_t)bm * K + (kt + 1) * GTK,
                                     &lA[nxt][0], (unsigned)K);
                tdm_load_tile_128x32(W_bf + (size_t)bn * K + (kt + 1) * GTK,
                                     &lB[nxt][0], (unsigned)K);
                __builtin_amdgcn_s_wait_tensorcnt(2);  // current pair landed
            } else {
                __builtin_amdgcn_s_wait_tensorcnt(0);
            }
        }
        __syncthreads();
#else
        {   // fallback: cooperative u32 copies of the bf16 tiles
            const unsigned* Asrc = (const unsigned*)(A_bf + (size_t)bm * K + kt * GTK);
            const unsigned* Wsrc = (const unsigned*)(W_bf + (size_t)bn * K + kt * GTK);
            unsigned* Adst = (unsigned*)&lA[cur][0];
            unsigned* Wdst = (unsigned*)&lB[cur][0];
            const int rowu = K >> 1;  // u32 per source row
            for (int i = tid; i < GTM * 16; i += 256)
                Adst[i] = Asrc[(size_t)(i >> 4) * rowu + (i & 15)];
            for (int i = tid; i < GTN * 16; i += 256)
                Wdst[i] = Wsrc[(size_t)(i >> 4) * rowu + (i & 15)];
            __syncthreads();
        }
#endif
        // B fragments (32x16 bf16): lane -> col n=l&15; ISA K interleave (same as A)
        BF16Frag bfrag[4];
#pragma unroll
        for (int nt = 0; nt < 4; ++nt) {
            const unsigned short* src = &lB[cur][(wn + nt * 16 + l15) * GTK];
#pragma unroll
            for (int j = 0; j < 8; ++j) {
                int kk = ((j >= 4) ? 16 : 0) + lhi * 8 + (j & 3) * 2;
                bfrag[nt].us[2 * j]     = src[kk];
                bfrag[nt].us[2 * j + 1] = src[kk + 1];
            }
        }
#pragma unroll
        for (int mt = 0; mt < 2; ++mt) {
            // A fragment (16x32 bf16): lane -> row m=l&15; ISA K interleave
            BF16Frag afrag;
            const unsigned short* src = &lA[cur][(wm + mt * 16 + l15) * GTK];
#pragma unroll
            for (int r = 0; r < 8; ++r) {
                int kk = ((r >= 4) ? 16 : 0) + lhi * 8 + (r & 3) * 2;
                afrag.us[2 * r]     = src[kk];
                afrag.us[2 * r + 1] = src[kk + 1];
            }
#pragma unroll
            for (int nt = 0; nt < 4; ++nt) {
                acc[mt][nt] = __builtin_amdgcn_wmma_f32_16x16x32_bf16(
                    false, afrag.bf, false, bfrag[nt].bf,
                    (short)0, acc[mt][nt], false, false);
            }
        }
        __syncthreads();
    }

    // D layout: lane -> col l&15; VGPR r -> row r + 8*(lane>>4)
#pragma unroll
    for (int mt = 0; mt < 2; ++mt)
#pragma unroll
        for (int nt = 0; nt < 4; ++nt) {
            int ncol = bn + wn + nt * 16 + l15;
            float bv = bias[ncol];
#pragma unroll
            for (int r = 0; r < 8; ++r) {
                int mrow = bm + wm + mt * 16 + r + 8 * lhi;
                Out[(size_t)mrow * N + ncol] = acc[mt][nt][r] + bv;
            }
        }
}

// ---------------------------------------------------------------------------
// Offset head + bilinear sampling (unchanged from round 1). One block per
// (b,g); thread p = spatial position. q/hs/xs token-major [B*196, 768].
// ---------------------------------------------------------------------------
__global__ __launch_bounds__(256)
void offset_sample(const float* __restrict__ q,
                   const float* __restrict__ hs,
                   const float* __restrict__ dww,
                   const float* __restrict__ dwb,
                   const float* __restrict__ lnw,
                   const float* __restrict__ lnb,
                   const float* __restrict__ pww,
                   float* __restrict__ xs)
{
    const int g = blockIdx.x;
    const int b = blockIdx.y;
    const int p = threadIdx.x;
    if (p >= NTOK) return;
    const int py = p / 14, px = p % 14;
    const int ch0 = g * CGDIM;
    const float* qb = q + (size_t)b * NTOK * CDIM;

    float s1 = 0.f, s2 = 0.f;
    for (int c = 0; c < CGDIM; ++c) {
        float t = dwb[c];
#pragma unroll
        for (int dy = -1; dy <= 1; ++dy)
#pragma unroll
            for (int dx = -1; dx <= 1; ++dx) {
                int yy = py + dy, xx = px + dx;
                if (yy >= 0 && yy < 14 && xx >= 0 && xx < 14)
                    t += dww[c * 9 + (dy + 1) * 3 + (dx + 1)] *
                         qb[(size_t)(yy * 14 + xx) * CDIM + ch0 + c];
            }
        s1 += t;
        s2 += t * t;
    }
    float mu   = s1 * (1.f / CGDIM);
    float var  = s2 * (1.f / CGDIM) - mu * mu;
    float rstd = rsqrtf(var + 1e-5f);

    float d0 = 0.f, d1 = 0.f;
    for (int c = 0; c < CGDIM; ++c) {
        float t = dwb[c];
#pragma unroll
        for (int dy = -1; dy <= 1; ++dy)
#pragma unroll
            for (int dx = -1; dx <= 1; ++dx) {
                int yy = py + dy, xx = px + dx;
                if (yy >= 0 && yy < 14 && xx >= 0 && xx < 14)
                    t += dww[c * 9 + (dy + 1) * 3 + (dx + 1)] *
                         qb[(size_t)(yy * 14 + xx) * CDIM + ch0 + c];
            }
        t = (t - mu) * rstd * lnw[c] + lnb[c];
        float ge = 0.5f * t * (1.f + erff(t * 0.70710678118654752f));
        d0 += pww[c] * ge;
        d1 += pww[CGDIM + c] * ge;
    }
    const float offy = tanhf(d0) * (1.f / 13.f);
    const float offx = tanhf(d1) * (1.f / 13.f);
    const float ry = ((py + 0.5f) / 14.f) * 2.f - 1.f;
    const float rx = ((px + 0.5f) / 14.f) * 2.f - 1.f;
    const float gx = ((offx + rx) + 1.f) * 0.5f * 13.f;
    const float gy = ((offy + ry) + 1.f) * 0.5f * 13.f;

    float x0 = floorf(gx), y0 = floorf(gy);
    float wx1 = gx - x0, wx0 = 1.f - wx1;
    float wy1 = gy - y0, wy0 = 1.f - wy1;
    int ix0 = (int)x0, iy0 = (int)y0;
    int ix1 = ix0 + 1, iy1 = iy0 + 1;

    float w00 = wx0 * wy0 * ((ix0 >= 0 && ix0 < 14 && iy0 >= 0 && iy0 < 14) ? 1.f : 0.f);
    float w10 = wx1 * wy0 * ((ix1 >= 0 && ix1 < 14 && iy0 >= 0 && iy0 < 14) ? 1.f : 0.f);
    float w01 = wx0 * wy1 * ((ix0 >= 0 && ix0 < 14 && iy1 >= 0 && iy1 < 14) ? 1.f : 0.f);
    float w11 = wx1 * wy1 * ((ix1 >= 0 && ix1 < 14 && iy1 >= 0 && iy1 < 14) ? 1.f : 0.f);

    int cx0 = iclampi(ix0, 0, 13), cx1 = iclampi(ix1, 0, 13);
    int cy0 = iclampi(iy0, 0, 13), cy1 = iclampi(iy1, 0, 13);

    const float* hb = hs + (size_t)b * NTOK * CDIM + ch0;
    const size_t i00 = (size_t)(cy0 * 14 + cx0) * CDIM;
    const size_t i10 = (size_t)(cy0 * 14 + cx1) * CDIM;
    const size_t i01 = (size_t)(cy1 * 14 + cx0) * CDIM;
    const size_t i11 = (size_t)(cy1 * 14 + cx1) * CDIM;
    float* xo = xs + (size_t)(b * NTOK + p) * CDIM + ch0;
    for (int c = 0; c < CGDIM; ++c)
        xo[c] = w00 * hb[i00 + c] + w10 * hb[i10 + c] +
                w01 * hb[i01 + c] + w11 * hb[i11 + c];
}

// ---------------------------------------------------------------------------
// Attention (unchanged from round 1): block = (b, head, 16-row strip),
// 13 waves; WMMA scores -> LDS softmax -> attn writeback -> WMMA context.
// ---------------------------------------------------------------------------
__global__ __launch_bounds__(416)
void attn_kernel(const float* __restrict__ q,
                 const float* __restrict__ k,
                 const float* __restrict__ v,
                 float* __restrict__ attn_o,
                 float* __restrict__ ctx)
{
    __shared__ float          S[16][224];
    __shared__ unsigned short Pb[16][224];
    __shared__ float          rsum[16];

    const int strip = blockIdx.x;
    const int h     = blockIdx.y;
    const int b     = blockIdx.z;
    const int tid   = threadIdx.x;
    const int wave  = tid >> 5;
    const int lane  = tid & 31;
    const int l15   = lane & 15;
    const int lhi   = lane >> 4;
    const int row0  = strip * 16;
    const float scale = 0.125f;

    v8f sacc = {};
    {
        int m = row0 + l15; if (m > 195) m = 195;
        int n = wave * 16 + l15; if (n > 195) n = 195;
        const float* qrow = q + (size_t)(b * NTOK + m) * CDIM + h * DHEAD;
        const float* krow = k + (size_t)(b * NTOK + n) * CDIM + h * DHEAD;
#pragma unroll
        for (int kt = 0; kt < 2; ++kt) {
            BF16Frag af, bf;
#pragma unroll
            for (int r = 0; r < 8; ++r) {
                int kk = kt * 32 + ((r >= 4) ? 16 : 0) + lhi * 8 + (r & 3) * 2;
                af.us[2 * r]     = f2bf(qrow[kk]);
                af.us[2 * r + 1] = f2bf(qrow[kk + 1]);
            }
#pragma unroll
            for (int j = 0; j < 8; ++j) {
                int kk = kt * 32 + ((j >= 4) ? 16 : 0) + lhi * 8 + (j & 3) * 2;
                bf.us[2 * j]     = f2bf(krow[kk]);
                bf.us[2 * j + 1] = f2bf(krow[kk + 1]);
            }
            sacc = __builtin_amdgcn_wmma_f32_16x16x32_bf16(
                false, af.bf, false, bf.bf, (short)0, sacc, false, false);
        }
    }
#pragma unroll
    for (int r = 0; r < 8; ++r)
        S[r + 8 * lhi][wave * 16 + l15] = sacc[r];
    __syncthreads();

    if (tid < 16) {
        int r = tid;
        float mx = -3.4e38f;
        for (int n = 0; n < NTOK; ++n) mx = fmaxf(mx, S[r][n] * scale);
        float sum = 0.f;
        for (int n = 0; n < NTOK; ++n) {
            float e = __expf(S[r][n] * scale - mx);
            S[r][n] = e;
            sum += e;
        }
        rsum[r] = 1.f / sum;
    }
    __syncthreads();

    for (int i = tid; i < 16 * 224; i += 416) {
        int r = i / 224, n = i % 224;
        float pval = (n < NTOK) ? S[r][n] * rsum[r] : 0.f;
        Pb[r][n] = f2bf(pval);
        if (n < NTOK && row0 + r < NTOK)
            attn_o[(((size_t)b * NHEAD + h) * NTOK + row0 + r) * NTOK + n] = pval;
    }
    __syncthreads();

    if (wave < 4) {
        v8f oacc = {};
        const int d = wave * 16 + l15;
        for (int kt = 0; kt < 7; ++kt) {
            BF16Frag af, bf;
#pragma unroll
            for (int r = 0; r < 8; ++r) {
                int kk = kt * 32 + ((r >= 4) ? 16 : 0) + lhi * 8 + (r & 3) * 2;
                af.us[2 * r]     = Pb[l15][kk];
                af.us[2 * r + 1] = Pb[l15][kk + 1];
            }
#pragma unroll
            for (int j = 0; j < 8; ++j) {
                int kk = kt * 32 + ((j >= 4) ? 16 : 0) + lhi * 8 + (j & 3) * 2;
                int n0 = kk     > 195 ? 195 : kk;
                int n1 = kk + 1 > 195 ? 195 : kk + 1;
                bf.us[2 * j]     = f2bf(v[(size_t)(b * NTOK + n0) * CDIM + h * DHEAD + d]);
                bf.us[2 * j + 1] = f2bf(v[(size_t)(b * NTOK + n1) * CDIM + h * DHEAD + d]);
            }
            oacc = __builtin_amdgcn_wmma_f32_16x16x32_bf16(
                false, af.bf, false, bf.bf, (short)0, oacc, false, false);
        }
#pragma unroll
        for (int r = 0; r < 8; ++r) {
            int m = row0 + r + 8 * lhi;
            if (m < NTOK)
                ctx[(size_t)(b * NTOK + m) * CDIM + h * DHEAD + d] = oacc[r];
        }
    }
}

// ---------------------------------------------------------------------------
// Host-side orchestration (all on `stream`, graph-capture safe)
// ---------------------------------------------------------------------------
extern "C" void kernel_launch(void* const* d_in, const int* in_sizes, int n_in,
                              void* d_out, int out_size, void* d_ws, size_t ws_size,
                              hipStream_t stream) {
    const float* hs  = (const float*)d_in[0];
    const float* Wq  = (const float*)d_in[1];
    const float* bq  = (const float*)d_in[2];
    const float* dww = (const float*)d_in[3];
    const float* dwb = (const float*)d_in[4];
    const float* lnw = (const float*)d_in[5];
    const float* lnb = (const float*)d_in[6];
    const float* pww = (const float*)d_in[7];
    const float* Wk  = (const float*)d_in[8];
    const float* bk  = (const float*)d_in[9];
    const float* Wv  = (const float*)d_in[10];
    const float* bv  = (const float*)d_in[11];
    const float* Wp  = (const float*)d_in[12];
    const float* bp  = (const float*)d_in[13];
    const float* Wo  = (const float*)d_in[14];
    const float* bo  = (const float*)d_in[15];

    float* ws  = (float*)d_ws;
    float* qb  = ws;               // Q, token-major fp32
    float* xsb = ws + S1ELEM;      // sampled feats; reused as ctx
    float* kb  = ws + 2 * S1ELEM;  // K; reused as y1
    float* vb  = ws + 3 * S1ELEM;  // V
    float* ctx = xsb;
    float* y1  = kb;
    float* y   = (float*)d_out;
    float* att = (float*)d_out + S1ELEM;

    unsigned short* bfb    = (unsigned short*)(ws + 4 * S1ELEM);
    unsigned short* act_bf = bfb;                    // S1ELEM bf16
    unsigned short* wq_bf  = bfb + S1ELEM;
    unsigned short* wk_bf  = wq_bf + WELEM;
    unsigned short* wv_bf  = wk_bf + WELEM;
    unsigned short* wp_bf  = wv_bf + WELEM;
    unsigned short* wo_bf  = wp_bf + WELEM;

    const long  nw2 = (long)(WELEM / 2);
    const long  na2 = (long)(S1ELEM / 2);
    const dim3  cvtW((unsigned)((nw2 + 255) / 256));
    const dim3  cvtA((unsigned)((na2 + 255) / 256));
    dim3 gemm_grid(MTOK / GTM, CDIM / GTN);          // 196 x 6

    // one-time weight conversion to bf16
    to_bf16_kernel<<<cvtW, 256, 0, stream>>>(Wq, (unsigned*)wq_bf, nw2);
    to_bf16_kernel<<<cvtW, 256, 0, stream>>>(Wk, (unsigned*)wk_bf, nw2);
    to_bf16_kernel<<<cvtW, 256, 0, stream>>>(Wv, (unsigned*)wv_bf, nw2);
    to_bf16_kernel<<<cvtW, 256, 0, stream>>>(Wp, (unsigned*)wp_bf, nw2);
    to_bf16_kernel<<<cvtW, 256, 0, stream>>>(Wo, (unsigned*)wo_bf, nw2);

    // Q = hs @ Wq^T + bq
    to_bf16_kernel<<<cvtA, 256, 0, stream>>>(hs, (unsigned*)act_bf, na2);
    gemm_bias_bf16<<<gemm_grid, 256, 0, stream>>>(act_bf, wq_bf, bq, qb, MTOK, CDIM, CDIM);
    // offsets + bilinear sampling
    offset_sample<<<dim3(GDIM, BATCH), 256, 0, stream>>>(qb, hs, dww, dwb, lnw, lnb, pww, xsb);
    // K, V projections of sampled features
    to_bf16_kernel<<<cvtA, 256, 0, stream>>>(xsb, (unsigned*)act_bf, na2);
    gemm_bias_bf16<<<gemm_grid, 256, 0, stream>>>(act_bf, wk_bf, bk, kb, MTOK, CDIM, CDIM);
    gemm_bias_bf16<<<gemm_grid, 256, 0, stream>>>(act_bf, wv_bf, bv, vb, MTOK, CDIM, CDIM);
    // attention (writes attn matrix to d_out and context to ctx)
    attn_kernel<<<dim3(13, NHEAD, BATCH), 416, 0, stream>>>(qb, kb, vb, att, ctx);
    // output projections
    to_bf16_kernel<<<cvtA, 256, 0, stream>>>(ctx, (unsigned*)act_bf, na2);
    gemm_bias_bf16<<<gemm_grid, 256, 0, stream>>>(act_bf, wp_bf, bp, y1, MTOK, CDIM, CDIM);
    to_bf16_kernel<<<cvtA, 256, 0, stream>>>(y1, (unsigned*)act_bf, na2);
    gemm_bias_bf16<<<gemm_grid, 256, 0, stream>>>(act_bf, wo_bf, bo, y, MTOK, CDIM, CDIM);
}
